// ResonantMultiheadAttention_12326556139612
// MI455X (gfx1250) — matrix operand order repaired and checked by reading, not compiled, not run on hardware
//
#include <hip/hip_runtime.h>
#include <hip/hip_bf16.h>

#define BDIM 2
#define TDIM 2048
#define DDIM 1024
#define HDS  16
#define HD   64
#define BT   (BDIM * TDIM)
#define NKT  (TDIM / 64)   // 32 k-tiles in attention

typedef __attribute__((ext_vector_type(16))) __bf16          v16bf;
typedef __attribute__((ext_vector_type(8)))  __bf16          v8bf;
typedef __attribute__((ext_vector_type(8)))  float           v8f;
typedef __attribute__((ext_vector_type(8)))  unsigned short  u16x8;

__device__ __forceinline__ unsigned short f2bf(float f) {
    unsigned int u = __builtin_bit_cast(unsigned int, f);
    u += 0x7fffu + ((u >> 16) & 1u);           // round-to-nearest-even
    return (unsigned short)(u >> 16);
}

// ---- CDNA5 async copy: global -> LDS, 16B per lane, tracked by ASYNCcnt ----
__device__ __forceinline__ void async_b128(void* lds_dst, const void* gbase, unsigned goff) {
    unsigned l = (unsigned)(unsigned long long)lds_dst;
    asm volatile("global_load_async_to_lds_b128 %0, %1, %2"
                 :: "v"(l), "v"(goff), "s"(gbase) : "memory");
}
__device__ __forceinline__ void wait_async() {
    asm volatile("s_wait_asynccnt 0x0" ::: "memory");
}

// ---- WMMA fragment builders (CDNA5 ISA 7.12.2 layouts, wave32) -------------
__device__ __forceinline__ v16bf load_frag_a(const unsigned short* p, int ld, int lane) {
    const unsigned short* row = p + (lane & 15) * ld + ((lane >> 4) << 3);
    v8bf lo = *(const v8bf*)(row);
    v8bf hi = *(const v8bf*)(row + 16);
    return __builtin_shufflevector(lo, hi, 0, 1, 2, 3, 4, 5, 6, 7,
                                           8, 9, 10, 11, 12, 13, 14, 15);
}
__device__ __forceinline__ v16bf load_frag_bc(const unsigned short* p, int ld, int lane) {
    const unsigned short* row = p + (lane & 15) * ld + ((lane >> 4) << 3);
    v8bf lo = *(const v8bf*)(row);
    v8bf hi = *(const v8bf*)(row + 16);
    return __builtin_shufflevector(lo, hi, 0, 1, 2, 3, 4, 5, 6, 7,
                                           8, 9, 10, 11, 12, 13, 14, 15);
}
__device__ __forceinline__ v8f wmma_bf16(v16bf a, v16bf b, v8f c) {
    return __builtin_amdgcn_wmma_f32_16x16x32_bf16(false, a, false, b, (short)0, c, false, false);
}

// ---- K0: f32 -> bf16 TRANSPOSED weight conversion (one-time) ---------------
__global__ void __launch_bounds__(256) k_convert_t(const float* __restrict__ w0,
                                                   const float* __restrict__ w1,
                                                   const float* __restrict__ w2,
                                                   const float* __restrict__ w3,
                                                   unsigned short* __restrict__ o0,
                                                   unsigned short* __restrict__ o1,
                                                   unsigned short* __restrict__ o2,
                                                   unsigned short* __restrict__ o3) {
    const float* a = (blockIdx.z == 0) ? w0 : (blockIdx.z == 1) ? w1
                   : (blockIdx.z == 2) ? w2 : w3;
    unsigned short* o = (blockIdx.z == 0) ? o0 : (blockIdx.z == 1) ? o1
                      : (blockIdx.z == 2) ? o2 : o3;
    __shared__ float tile[32][33];
    int tx = threadIdx.x & 31, ty = threadIdx.x >> 5;
    int c0 = blockIdx.x * 32, r0 = blockIdx.y * 32;
#pragma unroll
    for (int i = 0; i < 32; i += 8)
        tile[ty + i][tx] = a[(size_t)(r0 + ty + i) * DDIM + c0 + tx];
    __syncthreads();
#pragma unroll
    for (int i = 0; i < 32; i += 8)
        o[(size_t)(c0 + ty + i) * DDIM + r0 + tx] = f2bf(tile[tx][ty + i]);
}

// ---- K1: LayerNorm, emit x and LN(x) as bf16 -------------------------------
__global__ void __launch_bounds__(256) k_layernorm(const float* __restrict__ x,
                                                   const float* __restrict__ gamma,
                                                   const float* __restrict__ beta,
                                                   unsigned short* __restrict__ xbf,
                                                   unsigned short* __restrict__ xnbf) {
    int row = blockIdx.x, tid = threadIdx.x;
    const float* xr = x + (size_t)row * DDIM;
    float v[4]; float s = 0.f;
#pragma unroll
    for (int i = 0; i < 4; ++i) { v[i] = xr[tid + i * 256]; s += v[i]; }
    __shared__ float red[256];
    red[tid] = s; __syncthreads();
    for (int off = 128; off > 0; off >>= 1) {
        if (tid < off) red[tid] += red[tid + off];
        __syncthreads();
    }
    float mu = red[0] * (1.0f / DDIM);
    __syncthreads();
    float s2 = 0.f;
#pragma unroll
    for (int i = 0; i < 4; ++i) { float d = v[i] - mu; s2 += d * d; }
    red[tid] = s2; __syncthreads();
    for (int off = 128; off > 0; off >>= 1) {
        if (tid < off) red[tid] += red[tid + off];
        __syncthreads();
    }
    float rstd = rsqrtf(red[0] * (1.0f / DDIM) + 1e-5f);
#pragma unroll
    for (int i = 0; i < 4; ++i) {
        int c = tid + i * 256;
        float xn = (v[i] - mu) * rstd * gamma[c] + beta[c];
        xnbf[(size_t)row * DDIM + c] = f2bf(xn);
        xbf [(size_t)row * DDIM + c] = f2bf(v[i]);
    }
}

#define LDA 40   // padded A-tile row stride (u16): 80B rows, 16B aligned
#define LDB 40   // padded K-major W-tile row stride
#define LD72 72  // padded stride for 64-wide attention tiles: 144B rows

// Arena offsets (u16 units) for the attention kernel
#define OFF_Q   0       // 128*72 = 9216  (dead after Q-frag build)
#define OFF_K0  9216    // 64*72 = 4608
#define OFF_V0  13824   // 64*72 = 4608
#define OFF_K1  0       // overlays dead sQ
#define OFF_V1  4608
#define OFF_P   18432   // 8 waves * 16*72 = 9216

// ---- K2: fused QKV projection GEMM + per-head L2-norm + gates --------------
// Double-buffered async pipeline: wait(i) -> barrier -> issue(i+1) -> compute(i)
__global__ void __launch_bounds__(256) k_qkv(const unsigned short* __restrict__ xbf,
                                             const unsigned short* __restrict__ xnbf,
                                             const unsigned short* __restrict__ wqT,
                                             const unsigned short* __restrict__ wkT,
                                             const unsigned short* __restrict__ wvT,
                                             const float* __restrict__ g_q,
                                             const float* __restrict__ g_k,
                                             unsigned short* __restrict__ qh,
                                             unsigned short* __restrict__ kh,
                                             unsigned short* __restrict__ vh,
                                             float* __restrict__ gqb,
                                             float* __restrict__ gkb) {
    int h = blockIdx.x, mt = blockIdx.y, mat = blockIdx.z;
    const unsigned short* A  = (mat == 2) ? xbf : xnbf;
    const unsigned short* WT = (mat == 0) ? wqT : (mat == 1 ? wkT : wvT);
    unsigned short* dst      = (mat == 0) ? qh : (mat == 1 ? kh : vh);
    const float* gvecp       = (mat == 0) ? g_q : g_k;
    float* gbuf              = (mat == 0) ? gqb : gkb;
    int m0 = mt * 128, nb = h * HD;
    __shared__ __align__(16) unsigned short lA[2][128 * LDA];
    __shared__ __align__(16) unsigned short lBt[2][64 * LDB];
    int tid = threadIdx.x, wave = tid >> 5, lane = tid & 31;
    int lh = lane >> 4, n = lane & 15;
    auto issue = [&](int it, int bufi) {
        int k0 = it * 32;
#pragma unroll
        for (int i = 0; i < 2; ++i) {
            int vi = i * 256 + tid;
            int r = vi >> 2, c8 = (vi & 3) * 8;
            async_b128(&lA[bufi][r * LDA + c8], A,
                       (unsigned)(((m0 + r) * DDIM + k0 + c8) * 2));
        }
        if (tid < 128) {
            int r = tid >> 1, c8 = (tid & 1) * 8;
            async_b128(&lBt[bufi][r * LDB + c8], WT,
                       (unsigned)(((nb + r) * DDIM + k0 + c8) * 2));
        }
    };
    v8f acc[4] = {};
    issue(0, 0);
    for (int it = 0; it < DDIM / 32; ++it) {
        int cur = it & 1;
        wait_async();            // tile `it` (issued last iteration) done
        __syncthreads();         // published; buf cur^1 no longer read
        if (it + 1 < DDIM / 32) issue(it + 1, cur ^ 1);  // overlaps compute
        // preload all fragments -> progressive s_wait_dscnt instead of 0
        v16bf a  = load_frag_a(&lA[cur][(wave * 16) * LDA], LDA, lane);
        v16bf b0 = load_frag_bc(&lBt[cur][0 * 16 * LDB], LDB, lane);
        v16bf b1 = load_frag_bc(&lBt[cur][1 * 16 * LDB], LDB, lane);
        v16bf b2 = load_frag_bc(&lBt[cur][2 * 16 * LDB], LDB, lane);
        v16bf b3 = load_frag_bc(&lBt[cur][3 * 16 * LDB], LDB, lane);
        acc[0] = wmma_bf16(a, b0, acc[0]);
        acc[1] = wmma_bf16(a, b1, acc[1]);
        acc[2] = wmma_bf16(a, b2, acc[2]);
        acc[3] = wmma_bf16(a, b3, acc[3]);
    }
    if (mat < 2) {
#pragma unroll
        for (int j = 0; j < 8; ++j) {
            float s = acc[0][j] * acc[0][j] + acc[1][j] * acc[1][j]
                    + acc[2][j] * acc[2][j] + acc[3][j] * acc[3][j];
            s += __shfl_xor(s, 1, 32); s += __shfl_xor(s, 2, 32);
            s += __shfl_xor(s, 4, 32); s += __shfl_xor(s, 8, 32);
            float inv = 1.0f / fmaxf(sqrtf(s), 1e-12f);
            acc[0][j] *= inv; acc[1][j] *= inv; acc[2][j] *= inv; acc[3][j] *= inv;
        }
        float gv0 = gvecp[n], gv1 = gvecp[16 + n], gv2 = gvecp[32 + n], gv3 = gvecp[48 + n];
#pragma unroll
        for (int j = 0; j < 8; ++j) {
            float gp = acc[0][j] * gv0 + acc[1][j] * gv1 + acc[2][j] * gv2 + acc[3][j] * gv3;
            gp += __shfl_xor(gp, 1, 32); gp += __shfl_xor(gp, 2, 32);
            gp += __shfl_xor(gp, 4, 32); gp += __shfl_xor(gp, 8, 32);
            if (n == 0) {
                int m = m0 + wave * 16 + j + lh * 8;
                int bb = m >> 11, tt = m & (TDIM - 1);
                gbuf[(size_t)(bb * HDS + h) * TDIM + tt] =
                    0.25f + 0.75f / (1.0f + __expf(-gp));
            }
        }
    }
#pragma unroll
    for (int t = 0; t < 4; ++t)
#pragma unroll
        for (int j = 0; j < 8; ++j) {
            int m = m0 + wave * 16 + j + lh * 8;
            int bb = m >> 11, tt = m & (TDIM - 1);
            dst[((size_t)(bb * HDS + h) * TDIM + tt) * HD + t * 16 + n] = f2bf(acc[t][j]);
        }
}

// ---- K3: resonant attention, double-buffered async K/V pipeline ------------
__global__ void __launch_bounds__(256) k_attn(const unsigned short* __restrict__ qh,
                                              const unsigned short* __restrict__ kh,
                                              const unsigned short* __restrict__ vh,
                                              const float* __restrict__ gqb,
                                              const float* __restrict__ gkb,
                                              unsigned short* __restrict__ fused) {
    int qt = blockIdx.x, h = blockIdx.y, b = blockIdx.z;
    size_t hbase = (size_t)(b * HDS + h) * TDIM;
    int q0 = qt * 128;
    __shared__ __align__(16) unsigned short arena[27648]; // 54KB
    __shared__ float sGq[128];
    __shared__ float sGk[2][64];
    int tid = threadIdx.x, wave = tid >> 5, lane = tid & 31;
    int lh = lane >> 4, n = lane & 15;
    const unsigned short* qbase = qh + (hbase + q0) * HD;  // contiguous 16KB
    const unsigned short* kbase = kh + hbase * HD;
    // ---- prologue: Q tile + K/V tile 0 ----
#pragma unroll
    for (int i = 0; i < 4; ++i) {
        int vi = i * 256 + tid;
        int r = vi >> 3, c8 = (vi & 7) * 8;
        async_b128(arena + OFF_Q + r * LD72 + c8, qbase, (unsigned)(vi * 16));
    }
#pragma unroll
    for (int i = 0; i < 2; ++i) {
        int vi = i * 256 + tid;
        int r = vi >> 3, c8 = (vi & 7) * 8;
        async_b128(arena + OFF_K0 + r * LD72 + c8, kbase, (unsigned)(vi * 16));
        u16x8 v = *(const u16x8*)(vh + (hbase + r) * HD + c8);
#pragma unroll
        for (int j = 0; j < 8; ++j) arena[OFF_V0 + (c8 + j) * LD72 + r] = v[j];
    }
    if (tid < 128) sGq[tid] = gqb[hbase + q0 + tid];
    if (tid < 64)  sGk[0][tid] = gkb[hbase + tid];
    wait_async();
    __syncthreads();
    // Q fragments / q-gates are loop-invariant (sQ region dead afterwards)
    const unsigned short* qrow = arena + OFF_Q + (wave * 16) * LD72;
    v16bf a0 = load_frag_a(qrow, LD72, lane);        // d 0..31
    v16bf a1 = load_frag_a(qrow + 32, LD72, lane);   // d 32..63
    float gqv[8];
#pragma unroll
    for (int j = 0; j < 8; ++j) gqv[j] = sGq[wave * 16 + j + lh * 8];
    unsigned short* sPw = arena + OFF_P + wave * 16 * LD72;
    v8f o[4] = {};
    for (int kt = 0; kt < NKT; ++kt) {
        int cur = kt & 1, nxt = cur ^ 1;
        bool hasNext = (kt + 1) < NKT;
        unsigned short* sKcur = arena + (cur ? OFF_K1 : OFF_K0);
        unsigned short* sVcur = arena + (cur ? OFF_V1 : OFF_V0);
        unsigned short* sKnxt = arena + (nxt ? OFF_K1 : OFF_K0);
        unsigned short* sVnxt = arena + (nxt ? OFF_V1 : OFF_V0);
        __syncthreads();   // publish tile kt everywhere; buf nxt free
        u16x8 vreg[2]; float gknxt = 0.f;
        int vr[2], vc[2];
        if (hasNext) {
            int k1 = (kt + 1) * 64;
#pragma unroll
            for (int i = 0; i < 2; ++i) {
                int vi = i * 256 + tid;
                int r = vi >> 3, c8 = (vi & 7) * 8;
                async_b128(sKnxt + r * LD72 + c8, kbase,
                           (unsigned)(k1 * HD * 2 + vi * 16));
                vr[i] = r; vc[i] = c8;
                vreg[i] = *(const u16x8*)(vh + (hbase + k1 + r) * HD + c8);
            }
            if (tid < 64) gknxt = gkb[hbase + k1 + tid];
        }
        // ---- S = Q @ K^T, resonance * gates -> P (bf16, per-wave LDS) ----
#pragma unroll
        for (int t = 0; t < 4; ++t) {
            v16bf bk0 = load_frag_bc(sKcur + (t * 16) * LD72, LD72, lane);
            v16bf bk1 = load_frag_bc(sKcur + (t * 16) * LD72 + 32, LD72, lane);
            v8f s = {};
            s = wmma_bf16(a0, bk0, s);
            s = wmma_bf16(a1, bk1, s);
            float gkv = sGk[cur][t * 16 + n];
#pragma unroll
            for (int j = 0; j < 8; ++j) {
                float rs = 0.5f + 0.5f * tanhf((s[j] - 0.27f) * 15.0f);
                sPw[(j + lh * 8) * LD72 + t * 16 + n] = f2bf(rs * gqv[j] * gkv);
            }
        }
        // V-tile scatter for kt+1 lands here: global latency hid under S-WMMAs
        if (hasNext) {
#pragma unroll
            for (int i = 0; i < 2; ++i)
#pragma unroll
                for (int j = 0; j < 8; ++j)
                    sVnxt[(vc[i] + j) * LD72 + vr[i]] = vreg[i][j];
            if (tid < 64) sGk[nxt][tid] = gknxt;
        }
        // ---- O += P @ V ----
        v16bf p0 = load_frag_a(sPw, LD72, lane);        // k' 0..31
        v16bf p1 = load_frag_a(sPw + 32, LD72, lane);   // k' 32..63
#pragma unroll
        for (int t = 0; t < 4; ++t) {
            v16bf bv0 = load_frag_bc(sVcur + (t * 16) * LD72, LD72, lane);
            v16bf bv1 = load_frag_bc(sVcur + (t * 16) * LD72 + 32, LD72, lane);
            o[t] = wmma_bf16(p0, bv0, o[t]);
            o[t] = wmma_bf16(p1, bv1, o[t]);
        }
        if (hasNext) wait_async();  // next tile's asyncs (overlapped) finish
    }
    // fused layout [B, T, D] (undoes the head transpose for the output GEMM)
#pragma unroll
    for (int t = 0; t < 4; ++t)
#pragma unroll
        for (int j = 0; j < 8; ++j) {
            int row = q0 + wave * 16 + j + lh * 8;
            fused[((size_t)(b * TDIM + row)) * DDIM + h * HD + t * 16 + n] = f2bf(o[t][j]);
        }
}

// ---- K4: output projection GEMM + bias, f32 out ----------------------------
__global__ void __launch_bounds__(256) k_out(const unsigned short* __restrict__ fused,
                                             const unsigned short* __restrict__ woT,
                                             const float* __restrict__ b_o,
                                             float* __restrict__ out) {
    int nt = blockIdx.x, mt = blockIdx.y;
    int m0 = mt * 128, nb = nt * 64;
    __shared__ __align__(16) unsigned short lA[2][128 * LDA];
    __shared__ __align__(16) unsigned short lBt[2][64 * LDB];
    int tid = threadIdx.x, wave = tid >> 5, lane = tid & 31;
    int lh = lane >> 4, n = lane & 15;
    auto issue = [&](int it, int bufi) {
        int k0 = it * 32;
#pragma unroll
        for (int i = 0; i < 2; ++i) {
            int vi = i * 256 + tid;
            int r = vi >> 2, c8 = (vi & 3) * 8;
            async_b128(&lA[bufi][r * LDA + c8], fused,
                       (unsigned)(((m0 + r) * DDIM + k0 + c8) * 2));
        }
        if (tid < 128) {
            int r = tid >> 1, c8 = (tid & 1) * 8;
            async_b128(&lBt[bufi][r * LDB + c8], woT,
                       (unsigned)(((nb + r) * DDIM + k0 + c8) * 2));
        }
    };
    v8f acc[4] = {};
    issue(0, 0);
    for (int it = 0; it < DDIM / 32; ++it) {
        int cur = it & 1;
        wait_async();
        __syncthreads();
        if (it + 1 < DDIM / 32) issue(it + 1, cur ^ 1);
        v16bf a  = load_frag_a(&lA[cur][(wave * 16) * LDA], LDA, lane);
        v16bf b0 = load_frag_bc(&lBt[cur][0 * 16 * LDB], LDB, lane);
        v16bf b1 = load_frag_bc(&lBt[cur][1 * 16 * LDB], LDB, lane);
        v16bf b2 = load_frag_bc(&lBt[cur][2 * 16 * LDB], LDB, lane);
        v16bf b3 = load_frag_bc(&lBt[cur][3 * 16 * LDB], LDB, lane);
        acc[0] = wmma_bf16(a, b0, acc[0]);
        acc[1] = wmma_bf16(a, b1, acc[1]);
        acc[2] = wmma_bf16(a, b2, acc[2]);
        acc[3] = wmma_bf16(a, b3, acc[3]);
    }
#pragma unroll
    for (int t = 0; t < 4; ++t)
#pragma unroll
        for (int j = 0; j < 8; ++j) {
            int m = m0 + wave * 16 + j + lh * 8;
            int c = nb + t * 16 + n;
            out[(size_t)m * DDIM + c] = acc[t][j] + b_o[c];
        }
}

extern "C" void kernel_launch(void* const* d_in, const int* in_sizes, int n_in,
                              void* d_out, int out_size, void* d_ws, size_t ws_size,
                              hipStream_t stream) {
    const float* x     = (const float*)d_in[0];
    const float* w_q   = (const float*)d_in[1];
    const float* w_k   = (const float*)d_in[2];
    const float* w_v   = (const float*)d_in[3];
    const float* g_q   = (const float*)d_in[4];
    const float* g_k   = (const float*)d_in[5];
    const float* w_o   = (const float*)d_in[6];
    const float* b_o   = (const float*)d_in[7];
    const float* gamma = (const float*)d_in[8];
    const float* beta  = (const float*)d_in[9];
    float* out = (float*)d_out;

    char* ws = (char*)d_ws;
    const size_t MB = 1024ull * 1024ull;
    unsigned short* wqT   = (unsigned short*)(ws + 0 * MB);
    unsigned short* wkT   = (unsigned short*)(ws + 2 * MB);
    unsigned short* wvT   = (unsigned short*)(ws + 4 * MB);
    unsigned short* woT   = (unsigned short*)(ws + 6 * MB);
    unsigned short* x_bf  = (unsigned short*)(ws + 8 * MB);
    unsigned short* xn_bf = (unsigned short*)(ws + 16 * MB);
    unsigned short* q_h   = (unsigned short*)(ws + 24 * MB);
    unsigned short* k_h   = (unsigned short*)(ws + 32 * MB);
    unsigned short* v_h   = (unsigned short*)(ws + 40 * MB);
    float*          gq_b  = (float*)(ws + 48 * MB);
    float*          gk_b  = (float*)(ws + 49 * MB);
    unsigned short* fused = (unsigned short*)(ws + 50 * MB);

    k_convert_t<<<dim3(32, 32, 4), 256, 0, stream>>>(
        w_q, w_k, w_v, w_o, wqT, wkT, wvT, woT);

    k_layernorm<<<BT, 256, 0, stream>>>(x, gamma, beta, x_bf, xn_bf);

    k_qkv<<<dim3(HDS, BT / 128, 3), 256, 0, stream>>>(
        x_bf, xn_bf, wqT, wkT, wvT, g_q, g_k, q_h, k_h, v_h, gq_b, gk_b);

    k_attn<<<dim3(TDIM / 128, HDS, BDIM), 256, 0, stream>>>(
        q_h, k_h, v_h, gq_b, gk_b, fused);

    k_out<<<dim3(DDIM / 64, BT / 128), 256, 0, stream>>>(fused, woT, b_o, out);
}
